// StructuredAttention_67121748902195
// MI455X (gfx1250) — matrix operand order repaired and checked by reading, not compiled, not run on hardware
//
#include <hip/hip_runtime.h>
#include <hip/hip_bf16.h>
#include <math.h>

// MI455X / gfx1250 structured-attention pipeline.
// All dense GEMMs run through v_wmma_f32_16x16x32_bf16 (wave32, fp32 acc).
// Operands that would be K-strided are pre-transposed into bf16 so every
// WMMA fragment load is a contiguous 16B global_load_b128.

typedef __bf16 bf16;
typedef __attribute__((ext_vector_type(16))) __bf16 v16bf;
typedef __attribute__((ext_vector_type(8)))  float  v8f;

#define Bd   8
#define Td   1024
#define Dd   1024
#define SEMd 512
#define STRd 512
#define KPAD 1056   // 1025 rounded up to a multiple of 32 (WMMA K-step)

#define EPI_NONE    0
#define EPI_TANH    1
#define EPI_RELU    2
#define EPI_EXPTANH 3   // exp(tanh(v)) with zeroed diagonal (A-matrix build)

__device__ __forceinline__ bf16 f2bf(float f) {
  union { float f; unsigned u; } x; x.f = f;
  unsigned r = x.u + 0x7FFFu + ((x.u >> 16) & 1u);   // round-to-nearest-even
  unsigned short h = (unsigned short)(r >> 16);
  bf16 o; __builtin_memcpy(&o, &h, 2);
  return o;
}

// ---------------------------------------------------------------------------
// Generic WMMA GEMM: C[b] = epi( A[b] (MxK) * W[b] (NxK)^T + bias ), bf16 in,
// fp32 accumulate, optional fp32 and/or bf16 stores. One 16x16 tile per wave,
// 8 waves (128 rows) per 256-thread block. K % 32 == 0, M % 128 == 0,
// N % 16 == 0.
// ---------------------------------------------------------------------------
__global__ __launch_bounds__(256) void k_gemm_bf16(
    const bf16* __restrict__ A, long sA, int lda,
    const bf16* __restrict__ W, long sW, int ldw,
    const float* __restrict__ bias,
    float* __restrict__ Cf, long sCf, int ldcf,
    bf16* __restrict__ Cb, long sCb, int ldcb,
    int M, int N, int K, int epi)
{
  const int lane = threadIdx.x & 31;
  const int wave = threadIdx.x >> 5;
  const int b    = blockIdx.z;
  const int m0   = (blockIdx.y * 8 + wave) * 16;
  const int n0   = blockIdx.x * 16;
  if (m0 >= M || n0 >= N) return;

  const bf16* Ab = A + (long)b * sA;
  const bf16* Wb = W + (long)b * sW;

  // A fragment (16x32 bf16): lane<16 -> row m0+lane, K halves {0..7,16..23};
  // lane>=16 -> same row group shifted by K+8 (ISA 16-bit A layout).
  const int aRow = m0 + (lane & 15);
  const int aK   = (lane >> 4) << 3;          // 0 or 8
  // B fragment (32x16 bf16): lane holds col n0+(lane&15), K halves matching A.
  const int wRow = n0 + (lane & 15);
  const int wK   = (lane >> 4) << 3;          // 0 or 8
  const bf16* ap = Ab + (long)aRow * lda + aK;
  const bf16* wp = Wb + (long)wRow * ldw + wK;

  v8f acc = {};
  for (int k = 0; k < K; k += 32) {
    v16bf af, wf;
    uint4* au = reinterpret_cast<uint4*>(&af);
    au[0] = *reinterpret_cast<const uint4*>(ap + k);        // K .. K+7
    au[1] = *reinterpret_cast<const uint4*>(ap + k + 16);   // K+16 .. K+23
    uint4* wu = reinterpret_cast<uint4*>(&wf);
    wu[0] = *reinterpret_cast<const uint4*>(wp + k);        // K .. K+7
    wu[1] = *reinterpret_cast<const uint4*>(wp + k + 16);   // K+16 .. K+23
    acc = __builtin_amdgcn_wmma_f32_16x16x32_bf16(false, af, false, wf,
                                                  (short)0, acc, false, false);
  }

  // C layout: VGPR r -> row m0 + r + (lane>=16 ? 8 : 0), col n0 + (lane&15)
  const int rowBase = m0 + ((lane >> 4) << 3);
  const int col     = n0 + (lane & 15);
  const float bv    = bias ? bias[col] : 0.0f;
#pragma unroll
  for (int r = 0; r < 8; ++r) {
    const int row = rowBase + r;
    float v = acc[r] + bv;
    if      (epi == EPI_TANH)    v = tanhf(v);
    else if (epi == EPI_RELU)    v = fmaxf(v, 0.0f);
    else if (epi == EPI_EXPTANH) v = (row == col) ? 0.0f : expf(tanhf(v));
    if (Cf) Cf[(long)b * sCf + (long)row * ldcf + col] = v;
    if (Cb) Cb[(long)b * sCb + (long)row * ldcb + col] = f2bf(v);
  }
}

// ---------------------------------------------------------------------------
// Helpers
// ---------------------------------------------------------------------------
__global__ void k_zero32(unsigned* __restrict__ p, long nwords) {
  long i = (long)blockIdx.x * blockDim.x + threadIdx.x;
  if (i < nwords) p[i] = 0u;
}

__global__ void k_cvt(const float* __restrict__ s, bf16* __restrict__ d, long n) {
  long i = (long)blockIdx.x * blockDim.x + threadIdx.x;
  if (i < n) d[i] = f2bf(s[i]);
}

// d[c, r] = s[r, c]  (produce transposed bf16 weight for contiguous K loads)
__global__ void k_cvt_T(const float* __restrict__ s, bf16* __restrict__ d,
                        int rows, int cols) {
  long i = (long)blockIdx.x * blockDim.x + threadIdx.x;
  if (i >= (long)rows * cols) return;
  int r = (int)(i / cols), c = (int)(i % cols);
  d[(long)c * rows + r] = f2bf(s[i]);
}

// Split interleaved fwd/bwd halves into semantic / structural views and fan
// sem into every bf16 consumer layout (finp cols 0..511, semT, ssrT cols 1..T).
__global__ void k_split(const float* __restrict__ in,
                        const float* __restrict__ exparam,
                        bf16* __restrict__ str_bf,   // (B,T,STR)
                        bf16* __restrict__ finp_bf,  // (B,T,3*SEM)
                        bf16* __restrict__ semT_bf,  // (B,SEM,T)
                        bf16* __restrict__ ssrT_bf)  // (B,SEM,KPAD)
{
  long i = (long)blockIdx.x * blockDim.x + threadIdx.x;
  if (i >= (long)Bd * Td * SEMd) return;
  int c = (int)(i & (SEMd - 1));
  int t = (int)((i >> 9) & (Td - 1));
  int b = (int)(i >> 19);
  const float* row = in + ((long)b * Td + t) * Dd;
  float semv = (c < 256) ? row[c]       : row[512 + (c - 256)];
  float strv = (c < 256) ? row[256 + c] : row[768 + (c - 256)];
  long bt = (long)b * Td + t;
  str_bf [bt * STRd + c]            = f2bf(strv);
  finp_bf[bt * (3 * SEMd) + c]      = f2bf(semv);
  semT_bf[((long)b * SEMd + c) * Td + t]        = f2bf(semv);
  ssrT_bf[((long)b * SEMd + c) * KPAD + (t + 1)] = f2bf(semv);
  if (t == 0) ssrT_bf[((long)b * SEMd + c) * KPAD] = f2bf(exparam[c]);
}

// f_i[b,t] = exp(tanh(str_v[b,t,:] . wfi)) — one wave32 per token.
__global__ void k_fi(const float* __restrict__ in,
                     const float* __restrict__ wfi,
                     float* __restrict__ f_i) {
  int w = blockIdx.x * (blockDim.x >> 5) + (threadIdx.x >> 5);
  int lane = threadIdx.x & 31;
  if (w >= Bd * Td) return;
  const float* row = in + (long)w * Dd;
  float s = 0.0f;
  for (int c = lane; c < STRd; c += 32) {
    float strv = (c < 256) ? row[256 + c] : row[768 + (c - 256)];
    s += strv * wfi[c];
  }
  for (int off = 16; off > 0; off >>= 1) s += __shfl_xor(s, off, 32);
  if (lane == 0) f_i[w] = expf(tanhf(s));
}

// colsum[b,j] = sum_i A[b,i,j]
__global__ void k_colsum(const float* __restrict__ Aexp, float* __restrict__ cs) {
  int j = blockIdx.x * blockDim.x + threadIdx.x;
  if (j >= Bd * Td) return;
  int b = j >> 10, col = j & (Td - 1);
  const float* Ab = Aexp + (long)b * Td * Td;
  float s = 0.0f;
  for (int i = 0; i < Td; ++i) s += Ab[(long)i * Td + col];
  cs[j] = s;
}

// Lbar = -A + diag(colsum), row 0 <- f_i ; Inv = I
__global__ void k_build(const float* __restrict__ Aexp,
                        const float* __restrict__ cs,
                        const float* __restrict__ f_i,
                        float* __restrict__ Lbar, float* __restrict__ Inv) {
  long idx = (long)blockIdx.x * blockDim.x + threadIdx.x;
  if (idx >= (long)Bd * Td * Td) return;
  int j = (int)(idx & (Td - 1));
  int i = (int)((idx >> 10) & (Td - 1));
  int b = (int)(idx >> 20);
  float a = Aexp[idx];
  float l;
  if (i == 0)      l = f_i[b * Td + j];
  else             l = -a + ((i == j) ? cs[b * Td + j] : 0.0f);
  Lbar[idx] = l;
  Inv[idx]  = (i == j) ? 1.0f : 0.0f;
}

// In-place augmented Gauss-Jordan: one workgroup (1024 threads) per batch,
// pivot rows staged in LDS. L/Inv pairs for all 8 batches total 64MB -> stays
// resident in the 192MB L2, so the per-step sweeps run at L2 bandwidth.
__global__ void __launch_bounds__(1024) k_gauss_jordan(float* __restrict__ Lb,
                                                       float* __restrict__ Vb) {
  const int b = blockIdx.x;
  float* L = Lb + (long)b * Td * Td;
  float* V = Vb + (long)b * Td * Td;
  __shared__ float pl[Td];
  __shared__ float pv[Td];
  const int tid = threadIdx.x;
  for (int k = 0; k < Td; ++k) {
    float pinv = 1.0f / L[(long)k * Td + k];
    float a = L[(long)k * Td + tid] * pinv;
    float c = V[(long)k * Td + tid] * pinv;
    pl[tid] = a; pv[tid] = c;
    L[(long)k * Td + tid] = a;
    V[(long)k * Td + tid] = c;
    __syncthreads();
    if (tid != k) {
      float f = L[(long)tid * Td + k];
      float* Li = L + (long)tid * Td;
      float* Vi = V + (long)tid * Td;
      for (int j = 0; j < Td; ++j) {
        Li[j] = fmaf(-f, pl[j], Li[j]);
        Vi[j] = fmaf(-f, pv[j], Vi[j]);
      }
    }
    __syncthreads();
  }
}

// Edge/root marginals: dx (bf16 for cinp GEMM) and df (fp32 output + bf16
// padded copy for pinp GEMM). df[b,t,k] = k==0 ? d0[b,t] : dx[b,k-1,t].
__global__ void k_marginals(const float* __restrict__ Aexp,
                            const float* __restrict__ Inv,
                            const float* __restrict__ f_i,
                            float* __restrict__ df_out,   // (B,T,T+1)
                            bf16*  __restrict__ dx_bf,    // (B,T,T)
                            bf16*  __restrict__ df_bf) {  // (B,T,KPAD)
  int j = blockIdx.x * 16 + threadIdx.x;
  int i = blockIdx.y * 16 + threadIdx.y;
  int b = blockIdx.z;
  const float* Ab = Aexp + (long)b * Td * Td;
  const float* Vb = Inv  + (long)b * Td * Td;
  float a  = Ab[(long)i * Td + j];
  float t1 = (j == 0) ? 0.0f : a * Vb[(long)j * Td + j];   // A_ij * Inv_jj
  float t2 = (i == 0) ? 0.0f : a * Vb[(long)j * Td + i];   // A_ij * Inv_ji
  float dx = t1 - t2;
  dx_bf[(long)b * Td * Td + (long)i * Td + j] = f2bf(dx);
  df_out[(long)b * Td * (Td + 1) + (long)j * (Td + 1) + (i + 1)] = dx;
  df_bf [(long)b * Td * KPAD     + (long)j * KPAD     + (i + 1)] = f2bf(dx);
  if (i == 0) {
    float d0 = f_i[b * Td + j] * Vb[(long)j * Td + 0];
    df_out[(long)b * Td * (Td + 1) + (long)j * (Td + 1)] = d0;
    df_bf [(long)b * Td * KPAD     + (long)j * KPAD]     = f2bf(d0);
  }
}

// ---------------------------------------------------------------------------
extern "C" void kernel_launch(void* const* d_in, const int* in_sizes, int n_in,
                              void* d_out, int out_size, void* d_ws, size_t ws_size,
                              hipStream_t stream) {
  const float* input   = (const float*)d_in[0];
  const float* Wtp     = (const float*)d_in[1];
  const float* btp     = (const float*)d_in[2];
  const float* Wtc     = (const float*)d_in[3];
  const float* btc     = (const float*)d_in[4];
  const float* wfi     = (const float*)d_in[5];
  const float* Wbil    = (const float*)d_in[6];
  const float* exparam = (const float*)d_in[7];
  const float* Wfz     = (const float*)d_in[8];
  const float* bfz     = (const float*)d_in[9];

  // ---- workspace carve (~212 MB), 256B aligned ----
  char* cur = (char*)d_ws;
  auto carve = [&](long bytes) -> void* {
    void* p = (void*)cur;
    cur += (bytes + 255) & ~255L;
    return p;
  };
  const long BT  = (long)Bd * Td;
  bf16* str_bf  = (bf16*)carve(BT * STRd * 2);
  bf16* semT_bf = (bf16*)carve((long)Bd * SEMd * Td * 2);
  bf16* ssrT_bf = (bf16*)carve((long)Bd * SEMd * KPAD * 2);
  bf16* tp_bf   = (bf16*)carve(BT * STRd * 2);
  bf16* tc_bf   = (bf16*)carve(BT * STRd * 2);
  bf16* tpw_bf  = (bf16*)carve(BT * STRd * 2);
  bf16* finp_bf = (bf16*)carve(BT * 3 * SEMd * 2);
  bf16* df_bf   = (bf16*)carve(BT * KPAD * 2);
  bf16* dx_bf   = (bf16*)carve(BT * Td * 2);
  bf16* Wtp_bf  = (bf16*)carve((long)STRd * STRd * 2);
  bf16* Wtc_bf  = (bf16*)carve((long)STRd * STRd * 2);
  bf16* WbilT_bf= (bf16*)carve((long)STRd * STRd * 2);
  bf16* Wfz_bf  = (bf16*)carve((long)SEMd * 3 * SEMd * 2);
  float* Aexp   = (float*)carve(BT * Td * 4);
  float* Lbar   = (float*)carve(BT * Td * 4);
  float* LLinv  = (float*)carve(BT * Td * 4);
  float* colsum = (float*)carve(BT * 4);
  float* f_i    = (float*)carve(BT * 4);

  float* out_f  = (float*)d_out;                 // (B,T,SEM)
  float* df_out = out_f + BT * SEMd;             // (B,T,T+1)

  // ---- zero the K-padded bf16 buffers (cols 1025..1055 must be 0) ----
  {
    long w1 = (long)Bd * SEMd * KPAD / 2;        // ssrT words
    long w2 = BT * KPAD / 2;                     // df_bf words
    k_zero32<<<(int)((w1 + 255) / 256), 256, 0, stream>>>((unsigned*)ssrT_bf, w1);
    k_zero32<<<(int)((w2 + 255) / 256), 256, 0, stream>>>((unsigned*)df_bf, w2);
  }

  // ---- weight conversion (bf16, K-contiguous layouts) ----
  {
    long n = (long)STRd * STRd;
    k_cvt  <<<(int)((n + 255) / 256), 256, 0, stream>>>(Wtp, Wtp_bf, n);
    k_cvt  <<<(int)((n + 255) / 256), 256, 0, stream>>>(Wtc, Wtc_bf, n);
    k_cvt_T<<<(int)((n + 255) / 256), 256, 0, stream>>>(Wbil, WbilT_bf, STRd, STRd);
    long nf = (long)SEMd * 3 * SEMd;
    k_cvt  <<<(int)((nf + 255) / 256), 256, 0, stream>>>(Wfz, Wfz_bf, nf);
  }

  // ---- split + sem fan-out, root scores ----
  {
    long n = BT * SEMd;
    k_split<<<(int)((n + 255) / 256), 256, 0, stream>>>(input, exparam, str_bf,
                                                        finp_bf, semT_bf, ssrT_bf);
    k_fi<<<(int)(BT / 8), 256, 0, stream>>>(input, wfi, f_i);
  }

  // ---- tp = tanh(str @ Wtp^T + btp), tc likewise (bf16 out) ----
  k_gemm_bf16<<<dim3(STRd / 16, (int)(BT / 128), 1), 256, 0, stream>>>(
      str_bf, 0, STRd, Wtp_bf, 0, STRd, btp,
      nullptr, 0, 0, tp_bf, 0, STRd, (int)BT, STRd, STRd, EPI_TANH);
  k_gemm_bf16<<<dim3(STRd / 16, (int)(BT / 128), 1), 256, 0, stream>>>(
      str_bf, 0, STRd, Wtc_bf, 0, STRd, btc,
      nullptr, 0, 0, tc_bf, 0, STRd, (int)BT, STRd, STRd, EPI_TANH);

  // ---- tpw = tp @ Wbil (via WbilT, bf16 out) ----
  k_gemm_bf16<<<dim3(STRd / 16, (int)(BT / 128), 1), 256, 0, stream>>>(
      tp_bf, 0, STRd, WbilT_bf, 0, STRd, nullptr,
      nullptr, 0, 0, tpw_bf, 0, STRd, (int)BT, STRd, STRd, EPI_NONE);

  // ---- A = exp(tanh(tpw @ tc^T)) with zero diagonal, batched over B ----
  k_gemm_bf16<<<dim3(Td / 16, Td / 128, Bd), 256, 0, stream>>>(
      tpw_bf, (long)Td * STRd, STRd, tc_bf, (long)Td * STRd, STRd, nullptr,
      Aexp, (long)Td * Td, Td, nullptr, 0, 0, Td, Td, STRd, EPI_EXPTANH);

  // ---- Laplacian assembly + inversion ----
  k_colsum<<<(int)(BT / 256), 256, 0, stream>>>(Aexp, colsum);
  {
    long n = BT * Td;
    k_build<<<(int)((n + 255) / 256), 256, 0, stream>>>(Aexp, colsum, f_i, Lbar, LLinv);
  }
  k_gauss_jordan<<<Bd, 1024, 0, stream>>>(Lbar, LLinv);

  // ---- marginals: dx, d0, df (fp32 output + bf16 GEMM copies) ----
  k_marginals<<<dim3(Td / 16, Td / 16, Bd), dim3(16, 16), 0, stream>>>(
      Aexp, LLinv, f_i, df_out, dx_bf, df_bf);

  // ---- pinp = df @ ssr (K padded to 1056) -> finp cols [512..1024) ----
  k_gemm_bf16<<<dim3(SEMd / 16, Td / 128, Bd), 256, 0, stream>>>(
      df_bf, (long)Td * KPAD, KPAD, ssrT_bf, (long)SEMd * KPAD, KPAD, nullptr,
      nullptr, 0, 0, finp_bf + SEMd, (long)Td * 3 * SEMd, 3 * SEMd,
      Td, SEMd, KPAD, EPI_NONE);

  // ---- cinp = dx @ sem_v -> finp cols [1024..1536) ----
  k_gemm_bf16<<<dim3(SEMd / 16, Td / 128, Bd), 256, 0, stream>>>(
      dx_bf, (long)Td * Td, Td, semT_bf, (long)SEMd * Td, Td, nullptr,
      nullptr, 0, 0, finp_bf + 2 * SEMd, (long)Td * 3 * SEMd, 3 * SEMd,
      Td, SEMd, Td, EPI_NONE);

  // ---- out = relu(finp @ Wfz^T + bfz), fp32 to d_out ----
  k_gemm_bf16<<<dim3(SEMd / 16, (int)(BT / 128), 1), 256, 0, stream>>>(
      finp_bf, 0, 3 * SEMd, Wfz_bf, 0, 3 * SEMd, bfz,
      out_f, 0, SEMd, nullptr, 0, 0, (int)BT, SEMd, 3 * SEMd, EPI_RELU);

  (void)in_sizes; (void)n_in; (void)out_size; (void)ws_size;
}
